// IPDoubleStreamBlockProcessor_37117107372189
// MI455X (gfx1250) — matrix operand order repaired and checked by reading, not compiled, not run on hardware
//
#include <hip/hip_runtime.h>

// ---------------------------------------------------------------------------
// Types for CDNA5 WMMA (gfx1250, wave32)
// ---------------------------------------------------------------------------
typedef __attribute__((ext_vector_type(16))) __bf16 v16bf;
typedef __attribute__((ext_vector_type(8)))  __bf16 v8bf;
typedef __attribute__((ext_vector_type(8)))  float  v8f;

#define HID 3072
#define MLPD 12288
#define NH 24
#define HD 128
#define LT 512
#define LI 2048
#define LL 2560   // LT + LI
#define CTX 4096
#define IPN 16
#define IPPAD 64  // padded key count for IP attention V

__device__ inline unsigned short f2bf_u16(float f) {
  union { float f; unsigned u; } a; a.f = f;
  unsigned u = a.u;
  unsigned r = u + 0x7FFFu + ((u >> 16) & 1u);   // round-to-nearest-even
  return (unsigned short)(r >> 16);
}

__device__ inline v8f v8f_zero() {
  v8f z;
#pragma unroll
  for (int i = 0; i < 8; ++i) z[i] = 0.f;
  return z;
}

// A-fragment (16x32 bf16, MxK): element i of v16bf:
// i<8 -> K = 8*half + i ; i>=8 -> K = 16 + 8*half + (i-8).
__device__ inline v16bf ld_frag_a(const unsigned short* p /*row base at k0*/, int half) {
  const v8bf a = *(const v8bf*)(p + 8 * half);
  const v8bf b = *(const v8bf*)(p + 16 + 8 * half);
  v16bf r;
#pragma unroll
  for (int i = 0; i < 8; ++i) { r[i] = a[i]; r[8 + i] = b[i]; }
  return r;
}

// B-fragment (32x16 bf16, KxN), stored as rows of B^T (N-major):
// i<8 -> K = 8*half + i ; i>=8 -> K = 16 + 8*half + (i-8).
__device__ inline v16bf ld_frag_b(const unsigned short* p /*row base at k0*/, int half) {
  const v8bf a = *(const v8bf*)(p + 8 * half);
  const v8bf b = *(const v8bf*)(p + 16 + 8 * half);
  v16bf r;
#pragma unroll
  for (int i = 0; i < 8; ++i) { r[i] = a[i]; r[8 + i] = b[i]; }
  return r;
}

__device__ inline float rmax16(float v) {
#pragma unroll
  for (int m = 8; m >= 1; m >>= 1) v = fmaxf(v, __shfl_xor(v, m, 32));
  return v;
}
__device__ inline float rsum16(float v) {
#pragma unroll
  for (int m = 8; m >= 1; m >>= 1) v += __shfl_xor(v, m, 32);
  return v;
}

// ---------------------------------------------------------------------------
// Elementwise / small kernels
// ---------------------------------------------------------------------------
__global__ void silu_kernel(const float* __restrict__ v, float* __restrict__ sv, int n) {
  int i = blockIdx.x * 256 + threadIdx.x;
  if (i < n) { float x = v[i]; sv[i] = x / (1.f + expf(-x)); }
}

// Modulation GEMV: out[n] = b[n] + sum_k sv[k] * w[k*N + n].  K-split x8 for BW.
__global__ __launch_bounds__(256) void gemv_part_kernel(const float* __restrict__ sv,
                                                        const float* __restrict__ w,
                                                        float* __restrict__ part,
                                                        int K, int N) {
  int n = blockIdx.x * 256 + threadIdx.x;
  int s = blockIdx.y;
  int kc = K / 8;
  int k0 = s * kc, k1 = k0 + kc;
  float acc = 0.f;
  for (int k = k0; k < k1; ++k) acc += sv[k] * w[(size_t)k * N + n];
  part[(size_t)s * N + n] = acc;
}
__global__ void gemv_reduce_kernel(const float* __restrict__ part, const float* __restrict__ b,
                                   float* __restrict__ out, int N) {
  int n = blockIdx.x * 256 + threadIdx.x;
  if (n >= N) return;
  float acc = b[n];
#pragma unroll
  for (int s = 0; s < 8; ++s) acc += part[(size_t)s * N + n];
  out[n] = acc;
}

// LayerNorm + modulation: y = (1 + mod[sc*HID + n]) * ln(x)[n] + mod[sh*HID + n]
__global__ __launch_bounds__(256) void ln_mod_kernel(const float* __restrict__ x,
                                                     const float* __restrict__ mod,
                                                     int sh_idx, int sc_idx,
                                                     float* __restrict__ y) {
  int row = blockIdx.x;
  const float* xr = x + (size_t)row * HID;
  float s = 0.f, s2 = 0.f;
  for (int i = threadIdx.x; i < HID; i += 256) { float v = xr[i]; s += v; s2 += v * v; }
#pragma unroll
  for (int m = 16; m >= 1; m >>= 1) { s += __shfl_xor(s, m, 32); s2 += __shfl_xor(s2, m, 32); }
  __shared__ float sh_s[8], sh_s2[8];
  int wv = threadIdx.x >> 5, ln = threadIdx.x & 31;
  if (ln == 0) { sh_s[wv] = s; sh_s2[wv] = s2; }
  __syncthreads();
  float ts = 0.f, ts2 = 0.f;
#pragma unroll
  for (int i = 0; i < 8; ++i) { ts += sh_s[i]; ts2 += sh_s2[i]; }
  float mean = ts * (1.f / HID);
  float var = ts2 * (1.f / HID) - mean * mean;
  float inv = rsqrtf(var + 1e-6f);
  const float* shv = mod + (size_t)sh_idx * HID;
  const float* scv = mod + (size_t)sc_idx * HID;
  float* yr = y + (size_t)row * HID;
  for (int i = threadIdx.x; i < HID; i += 256) {
    float v = (xr[i] - mean) * inv;
    yr[i] = (1.f + scv[i]) * v + shv[i];
  }
}

// f32 -> bf16 convert
__global__ void cvt_kernel(const float* __restrict__ in, unsigned short* __restrict__ out, int n) {
  int i = blockIdx.x * 256 + threadIdx.x;
  if (i < n) out[i] = f2bf_u16(in[i]);
}

// ip V: (16, HID) f32 -> (NH, HD, IPPAD) bf16 transposed, zero-padded keys
__global__ void ipv_transpose_kernel(const float* __restrict__ in,
                                     unsigned short* __restrict__ out) {
  int idx = blockIdx.x * 256 + threadIdx.x;     // NH*HD*IPPAD = 196608
  int key = idx & (IPPAD - 1);
  int d = (idx >> 6) & (HD - 1);
  int h = idx >> 13;
  float v = (key < IPN) ? in[(size_t)key * HID + h * HD + d] : 0.f;
  out[idx] = f2bf_u16(v);
}

// per-head u16 transpose: in (H, R, C) -> out (H, C, R), 32x32 LDS tiles
__global__ __launch_bounds__(256) void transpose_u16_kernel(const unsigned short* __restrict__ in,
                                                            unsigned short* __restrict__ out,
                                                            int R, int C) {
  __shared__ unsigned short tile[32][33];
  int h = blockIdx.z;
  int c0 = blockIdx.x * 32, r0 = blockIdx.y * 32;
  const unsigned short* ih = in + (size_t)h * R * C;
  unsigned short* oh = out + (size_t)h * R * C;
#pragma unroll
  for (int j = 0; j < 32; j += 8) {
    int r = r0 + threadIdx.y + j;
    tile[threadIdx.y + j][threadIdx.x] = ih[(size_t)r * C + c0 + threadIdx.x];
  }
  __syncthreads();
#pragma unroll
  for (int j = 0; j < 32; j += 8) {
    int c = c0 + threadIdx.y + j;
    oh[(size_t)c * R + r0 + threadIdx.x] = tile[threadIdx.x][threadIdx.y + j];
  }
}

// QKV postprocess: RMS-norm q/k (per head, over D=128) * scale, RoPE q/k,
// write bf16 Q/K/V in (H, LL, D) head-major layout; img stream also writes
// un-roped normalized q for the IP attention.
__global__ __launch_bounds__(128) void qkv_post_kernel(const float* __restrict__ qkv,
                                                       const float* __restrict__ qs,
                                                       const float* __restrict__ ks,
                                                       const float* __restrict__ pe,
                                                       int row_off, int Lq_img_mode,
                                                       unsigned short* __restrict__ Qr,
                                                       unsigned short* __restrict__ Kr,
                                                       unsigned short* __restrict__ Vr,
                                                       unsigned short* __restrict__ Qimg) {
  int l = blockIdx.x, h = blockIdx.y, d = threadIdx.x;
  const float* base = qkv + (size_t)l * (3 * HID) + (size_t)h * HD + d;
  float q = base[0], k = base[HID], v = base[2 * HID];

  __shared__ float red[4];
  float ss = q * q;
#pragma unroll
  for (int m = 16; m >= 1; m >>= 1) ss += __shfl_xor(ss, m, 32);
  if ((threadIdx.x & 31) == 0) red[threadIdx.x >> 5] = ss;
  __syncthreads();
  float t = red[0] + red[1] + red[2] + red[3];
  __syncthreads();
  float qn = q * rsqrtf(t * (1.f / HD) + 1e-6f) * qs[d];
  ss = k * k;
#pragma unroll
  for (int m = 16; m >= 1; m >>= 1) ss += __shfl_xor(ss, m, 32);
  if ((threadIdx.x & 31) == 0) red[threadIdx.x >> 5] = ss;
  __syncthreads();
  t = red[0] + red[1] + red[2] + red[3];
  __syncthreads();
  float kn = k * rsqrtf(t * (1.f / HD) + 1e-6f) * ks[d];

  __shared__ float sq[HD], sk[HD];
  sq[d] = qn; sk[d] = kn;
  __syncthreads();

  int lcat = row_off + l;
  int i = d >> 1, j = d & 1;
  const float* pep = pe + (size_t)lcat * 256 + i * 4 + j * 2;  // pe (LL,64,2,2)
  float x0 = sq[2 * i], x1 = sq[2 * i + 1];
  float oq = pep[0] * x0 + pep[1] * x1;
  x0 = sk[2 * i]; x1 = sk[2 * i + 1];
  float ok = pep[0] * x0 + pep[1] * x1;

  size_t o = ((size_t)h * LL + lcat) * HD + d;
  Qr[o] = f2bf_u16(oq);
  Kr[o] = f2bf_u16(ok);
  Vr[o] = f2bf_u16(v);
  if (Lq_img_mode) Qimg[((size_t)h * LI + l) * HD + d] = f2bf_u16(qn);
}

// ---------------------------------------------------------------------------
// Generic bf16-WMMA GEMM:  C(M,N) = epilogue( A(M,K) * B(K,N) + bias )
// Block tile 128x256, BK=64, 8 waves as 2(m) x 4(n) -> 64x64 per wave.
// Register-prefetch pipeline: tile k+1 loads overlap tile k WMMAs.
// mode 0: acc+bias | mode 1: gelu_tanh(acc+bias) | mode 2: res + gate*(acc+bias)
// mode 3: res + gate*(acc+bias) + escale[0]*extra   (fused IP-attention add)
// ---------------------------------------------------------------------------
#define BM 128
#define BN 256
#define BK 64

__global__ __launch_bounds__(256) void gemm_kernel(const float* __restrict__ A,
                                                   const float* __restrict__ B,
                                                   const float* __restrict__ bias,
                                                   const float* __restrict__ res,
                                                   const float* __restrict__ gate,
                                                   const float* __restrict__ extra,
                                                   const float* __restrict__ escale,
                                                   float* __restrict__ C,
                                                   int M, int N, int K, int mode) {
  __shared__ alignas(32) unsigned short As[BM * BK];  // [m][k]  16 KB
  __shared__ alignas(32) unsigned short Bs[BN * BK];  // [n][k]  32 KB (B transposed)
  int bn = blockIdx.x * BN;
  int bm = blockIdx.y * BM;
  int tid = threadIdx.x;
  int lane = tid & 31, wv = tid >> 5;
  int wm = wv >> 2, wn = wv & 3;     // 2 x 4 wave grid -> wave tile 64 x 64
  int col = lane & 15, half = lane >> 4;

  v8f acc[4][4];
#pragma unroll
  for (int a = 0; a < 4; ++a)
#pragma unroll
    for (int b = 0; b < 4; ++b) acc[a][b] = v8f_zero();

  // per-thread prefetch registers: A 8 x float4, B 8 slots x 2 rows x float4
  float4 pa[8], pb0[8], pb1[8];

  auto gload = [&](int k0) {
#pragma unroll
    for (int i = 0; i < 8; ++i) {               // A: 128 x 64 f32
      int idx = tid + i * 256;
      int r = idx >> 4, c4 = (idx & 15) << 2;
      int gm = bm + r;
      gm = (gm < M) ? gm : (M - 1);
      pa[i] = *(const float4*)(A + (size_t)gm * K + k0 + c4);
    }
#pragma unroll
    for (int i = 0; i < 8; ++i) {               // B: 64 x 256 f32, 2 rows/slot
      int idx = tid + i * 256;
      int kp = idx >> 6, c4 = (idx & 63) << 2;  // kp: k-pair 0..31
      const float* bp = B + (size_t)(k0 + 2 * kp) * N + bn + c4;
      pb0[i] = *(const float4*)bp;
      pb1[i] = *(const float4*)(bp + N);
    }
  };

  gload(0);
  for (int k0 = 0; k0 < K; k0 += BK) {
    // registers -> LDS (bf16)
#pragma unroll
    for (int i = 0; i < 8; ++i) {
      int idx = tid + i * 256;
      int r = idx >> 4, c4 = (idx & 15) << 2;
      unsigned lo = (unsigned)f2bf_u16(pa[i].x) | ((unsigned)f2bf_u16(pa[i].y) << 16);
      unsigned hi = (unsigned)f2bf_u16(pa[i].z) | ((unsigned)f2bf_u16(pa[i].w) << 16);
      *(uint2*)(&As[r * BK + c4]) = make_uint2(lo, hi);
    }
#pragma unroll
    for (int i = 0; i < 8; ++i) {
      int idx = tid + i * 256;
      int kp = idx >> 6, c4 = (idx & 63) << 2;
      *(unsigned*)&Bs[(c4 + 0) * BK + 2 * kp] =
          (unsigned)f2bf_u16(pb0[i].x) | ((unsigned)f2bf_u16(pb1[i].x) << 16);
      *(unsigned*)&Bs[(c4 + 1) * BK + 2 * kp] =
          (unsigned)f2bf_u16(pb0[i].y) | ((unsigned)f2bf_u16(pb1[i].y) << 16);
      *(unsigned*)&Bs[(c4 + 2) * BK + 2 * kp] =
          (unsigned)f2bf_u16(pb0[i].z) | ((unsigned)f2bf_u16(pb1[i].z) << 16);
      *(unsigned*)&Bs[(c4 + 3) * BK + 2 * kp] =
          (unsigned)f2bf_u16(pb0[i].w) | ((unsigned)f2bf_u16(pb1[i].w) << 16);
    }
    __syncthreads();

    if (k0 + BK < K) gload(k0 + BK);  // prefetch next tile; overlaps WMMAs below

#pragma unroll
    for (int kc = 0; kc < BK; kc += 32) {
      v16bf af[4], bf[4];
#pragma unroll
      for (int t = 0; t < 4; ++t)
        af[t] = ld_frag_a(&As[(wm * 64 + t * 16 + col) * BK + kc], half);
#pragma unroll
      for (int t = 0; t < 4; ++t)
        bf[t] = ld_frag_b(&Bs[(wn * 64 + t * 16 + col) * BK + kc], half);
#pragma unroll
      for (int a = 0; a < 4; ++a)
#pragma unroll
        for (int b = 0; b < 4; ++b)
          acc[a][b] = __builtin_amdgcn_wmma_f32_16x16x32_bf16(
              false, af[a], false, bf[b], (short)0, acc[a][b], false, false);
    }
    __syncthreads();
  }

  float esc = (mode == 3) ? escale[0] : 0.f;
#pragma unroll
  for (int a = 0; a < 4; ++a)
#pragma unroll
    for (int b = 0; b < 4; ++b) {
      int nn = bn + wn * 64 + b * 16 + col;
      float bv = bias[nn];
#pragma unroll
      for (int i = 0; i < 8; ++i) {
        int mm = bm + wm * 64 + a * 16 + i + 8 * half;
        if (mm >= M) continue;
        float v = acc[a][b][i] + bv;
        if (mode == 1) {
          float x = v;
          v = 0.5f * x * (1.f + tanhf(0.7978845608028654f * (x + 0.044715f * x * x * x)));
        } else if (mode == 2) {
          v = res[(size_t)mm * N + nn] + gate[nn] * v;
        } else if (mode == 3) {
          v = res[(size_t)mm * N + nn] + gate[nn] * v + esc * extra[(size_t)mm * N + nn];
        }
        C[(size_t)mm * N + nn] = v;
      }
    }
}

// ---------------------------------------------------------------------------
// Flash attention (wave32 WMMA). 4 independent waves / block, 16 q rows each.
// Q (H,Lq,HD) bf16 row-major; K (rows=keys) bf16 via k_rs/k_hs;
// V pre-TRANSPOSED bf16: head base + d*v_ls + key  (v_ls >= padded Lk).
// No block barriers needed: P round-trip LDS is wave-private.
// ---------------------------------------------------------------------------
__global__ __launch_bounds__(128) void attn_kernel(const unsigned short* __restrict__ Q,
                                                   long q_hs,
                                                   const unsigned short* __restrict__ Kp,
                                                   int k_rs, long k_hs,
                                                   const unsigned short* __restrict__ Vt,
                                                   int v_ls, long v_hs,
                                                   float* __restrict__ Out,
                                                   int out_rs, int out_head_stride,
                                                   int Lq, int Lk, float scale) {
  __shared__ alignas(32) unsigned short Pw[4 * 16 * 64]; // per-wave P tiles
  int h = blockIdx.y;
  int q0blk = blockIdx.x * 64;
  int tid = threadIdx.x, lane = tid & 31, wv = tid >> 5;
  int col = lane & 15, half = lane >> 4;

  const unsigned short* Qh = Q + (size_t)h * q_hs;
  const unsigned short* Kh = Kp + (size_t)h * k_hs;
  const unsigned short* Vh = Vt + (size_t)h * v_hs;

  int qrow = q0blk + wv * 16 + col;
  const unsigned short* qp = Qh + (size_t)qrow * HD;
  v16bf qf[4];
#pragma unroll
  for (int c = 0; c < 4; ++c) qf[c] = ld_frag_a(qp + 32 * c, half);

  v8f o[8];
  float mi[8], li[8];
#pragma unroll
  for (int i = 0; i < 8; ++i) { o[i] = v8f_zero(); mi[i] = -3e38f; li[i] = 0.f; }

  unsigned short* Pme = &Pw[wv * 16 * 64];
  int nkt = (Lk + 63) >> 6;

  for (int kt = 0; kt < nkt; ++kt) {
    int kbase = kt * 64;

    // S = Q * K^T   (B-fragments read straight from K rows)
    v8f st[4];
#pragma unroll
    for (int t = 0; t < 4; ++t) {
      st[t] = v8f_zero();
      int krow = kbase + t * 16 + col;
      int krc = (krow < Lk) ? krow : 0;
      const unsigned short* kp = Kh + (size_t)krc * k_rs;
#pragma unroll
      for (int c = 0; c < 4; ++c) {
        v16bf kf = ld_frag_b(kp + 32 * c, half);
        st[t] = __builtin_amdgcn_wmma_f32_16x16x32_bf16(
            false, qf[c], false, kf, (short)0, st[t], false, false);
      }
    }

    // online softmax over the 64 keys of this tile
#pragma unroll
    for (int i = 0; i < 8; ++i) {
      float mx = -3e38f;
#pragma unroll
      for (int t = 0; t < 4; ++t) {
        int kg = kbase + t * 16 + col;
        float v = (kg < Lk) ? st[t][i] * scale : -3e38f;
        st[t][i] = v;
        mx = fmaxf(mx, v);
      }
      mx = rmax16(mx);
      float mn = fmaxf(mi[i], mx);
      float alpha = __expf(mi[i] - mn);
      mi[i] = mn;
      float rs = 0.f;
#pragma unroll
      for (int t = 0; t < 4; ++t) {
        float p = __expf(st[t][i] - mn);
        st[t][i] = p;
        rs += p;
      }
      rs = rsum16(rs);
      li[i] = li[i] * alpha + rs;
#pragma unroll
      for (int dt = 0; dt < 8; ++dt) o[dt][i] *= alpha;
      int r = i + 8 * half;
#pragma unroll
      for (int t = 0; t < 4; ++t) Pme[r * 64 + t * 16 + col] = f2bf_u16(st[t][i]);
    }

    // O += P * V  (A-fragments from Pme; V B-fragments direct from transposed global)
#pragma unroll
    for (int c = 0; c < 2; ++c) {
      v16bf pf = ld_frag_a(&Pme[col * 64 + 32 * c], half);
#pragma unroll
      for (int dt = 0; dt < 8; ++dt) {
        v16bf vf = ld_frag_b(Vh + (size_t)(dt * 16 + col) * v_ls + kbase + 32 * c, half);
        o[dt] = __builtin_amdgcn_wmma_f32_16x16x32_bf16(
            false, pf, false, vf, (short)0, o[dt], false, false);
      }
    }
  }

  float* Oh = Out + (size_t)h * out_head_stride;
#pragma unroll
  for (int i = 0; i < 8; ++i) {
    int r = q0blk + wv * 16 + i + 8 * half;
    if (r >= Lq) continue;
    float invl = 1.f / li[i];
#pragma unroll
    for (int dt = 0; dt < 8; ++dt)
      Oh[(size_t)r * out_rs + dt * 16 + col] = o[dt][i] * invl;
  }
}

// ---------------------------------------------------------------------------
// Host launcher
// ---------------------------------------------------------------------------
extern "C" void kernel_launch(void* const* d_in, const int* in_sizes, int n_in,
                              void* d_out, int out_size, void* d_ws, size_t ws_size,
                              hipStream_t stream) {
  (void)in_sizes; (void)n_in; (void)out_size; (void)ws_size;
  const float* img        = (const float*)d_in[0];
  const float* txt        = (const float*)d_in[1];
  const float* vec        = (const float*)d_in[2];
  const float* pe         = (const float*)d_in[3];
  const float* image_proj = (const float*)d_in[4];
  const float* img_mod_w  = (const float*)d_in[5];
  const float* img_mod_b  = (const float*)d_in[6];
  const float* txt_mod_w  = (const float*)d_in[7];
  const float* txt_mod_b  = (const float*)d_in[8];
  const float* img_qkv_w  = (const float*)d_in[9];
  const float* img_qkv_b  = (const float*)d_in[10];
  const float* txt_qkv_w  = (const float*)d_in[11];
  const float* txt_qkv_b  = (const float*)d_in[12];
  const float* img_q_scale = (const float*)d_in[13];
  const float* img_k_scale = (const float*)d_in[14];
  const float* txt_q_scale = (const float*)d_in[15];
  const float* txt_k_scale = (const float*)d_in[16];
  const float* img_proj_w = (const float*)d_in[17];
  const float* img_proj_b = (const float*)d_in[18];
  const float* txt_proj_w = (const float*)d_in[19];
  const float* txt_proj_b = (const float*)d_in[20];
  const float* img_mlp_w1 = (const float*)d_in[21];
  const float* img_mlp_b1 = (const float*)d_in[22];
  const float* img_mlp_w2 = (const float*)d_in[23];
  const float* img_mlp_b2 = (const float*)d_in[24];
  const float* txt_mlp_w1 = (const float*)d_in[25];
  const float* txt_mlp_b1 = (const float*)d_in[26];
  const float* txt_mlp_w2 = (const float*)d_in[27];
  const float* txt_mlp_b2 = (const float*)d_in[28];
  const float* ip_k_w     = (const float*)d_in[29];
  const float* ip_k_b     = (const float*)d_in[30];
  const float* ip_v_w     = (const float*)d_in[31];
  const float* ip_v_b     = (const float*)d_in[32];
  const float* ip_scale   = (const float*)d_in[33];

  float* OUT = (float*)d_out;              // img: [0, LI*HID), txt: [LI*HID, ...)

  char* base = (char*)d_ws;
  size_t off = 0;
  auto alloc = [&](size_t bytes) -> void* {
    void* p = base + off;
    off = (off + bytes + 255) & ~(size_t)255;
    return p;
  };

  float* sv    = (float*)alloc(HID * 4);
  float* part  = (float*)alloc((size_t)8 * 6 * HID * 4);
  float* modI  = (float*)alloc((size_t)6 * HID * 4);
  float* modT  = (float*)alloc((size_t)6 * HID * 4);
  float* lnI   = (float*)alloc((size_t)LI * HID * 4);
  float* lnT   = (float*)alloc((size_t)LT * HID * 4);
  float* qkvI  = (float*)alloc((size_t)LI * 3 * HID * 4);
  float* qkvT  = (float*)alloc((size_t)LT * 3 * HID * 4);
  unsigned short* Qr   = (unsigned short*)alloc((size_t)NH * LL * HD * 2);
  unsigned short* Kr   = (unsigned short*)alloc((size_t)NH * LL * HD * 2);
  unsigned short* Vr   = (unsigned short*)alloc((size_t)NH * LL * HD * 2);
  unsigned short* Vtr  = (unsigned short*)alloc((size_t)NH * HD * LL * 2);
  unsigned short* Qimg = (unsigned short*)alloc((size_t)NH * LI * HD * 2);
  float* attn1 = (float*)alloc((size_t)LL * HID * 4);
  float* img1  = (float*)alloc((size_t)LI * HID * 4);
  float* txt1  = (float*)alloc((size_t)LT * HID * 4);
  float* hI    = (float*)alloc((size_t)LI * MLPD * 4);
  float* hT    = (float*)alloc((size_t)LT * MLPD * 4);
  float* ipk   = (float*)alloc((size_t)IPN * HID * 4);
  float* ipv   = (float*)alloc((size_t)IPN * HID * 4);
  unsigned short* ipkb = (unsigned short*)alloc((size_t)IPN * HID * 2);
  unsigned short* ipvt = (unsigned short*)alloc((size_t)NH * HD * IPPAD * 2);
  float* ipattn = (float*)alloc((size_t)LI * HID * 4);

  const float attn_scale = 0.08838834764831845f;  // 1/sqrt(128)

  // 1) silu(vec)
  silu_kernel<<<(HID + 255) / 256, 256, 0, stream>>>(vec, sv, HID);

  // 2) modulation GEMVs
  {
    dim3 g(6 * HID / 256, 8);
    gemv_part_kernel<<<g, 256, 0, stream>>>(sv, img_mod_w, part, HID, 6 * HID);
    gemv_reduce_kernel<<<6 * HID / 256, 256, 0, stream>>>(part, img_mod_b, modI, 6 * HID);
    gemv_part_kernel<<<g, 256, 0, stream>>>(sv, txt_mod_w, part, HID, 6 * HID);
    gemv_reduce_kernel<<<6 * HID / 256, 256, 0, stream>>>(part, txt_mod_b, modT, 6 * HID);
  }

  // 3) LN + mod (pre-attention): sh=0, sc=1
  ln_mod_kernel<<<LI, 256, 0, stream>>>(img, modI, 0, 1, lnI);
  ln_mod_kernel<<<LT, 256, 0, stream>>>(txt, modT, 0, 1, lnT);

  // 4) QKV GEMMs
  {
    dim3 gI(3 * HID / BN, LI / BM);
    gemm_kernel<<<gI, 256, 0, stream>>>(lnI, img_qkv_w, img_qkv_b, nullptr, nullptr,
                                        nullptr, nullptr, qkvI, LI, 3 * HID, HID, 0);
    dim3 gT(3 * HID / BN, LT / BM);
    gemm_kernel<<<gT, 256, 0, stream>>>(lnT, txt_qkv_w, txt_qkv_b, nullptr, nullptr,
                                        nullptr, nullptr, qkvT, LT, 3 * HID, HID, 0);
  }

  // 5) QKV post: RMS + RoPE -> bf16 head-major (txt rows first, img after)
  {
    dim3 gT(LT, NH);
    qkv_post_kernel<<<gT, 128, 0, stream>>>(qkvT, txt_q_scale, txt_k_scale, pe, 0, 0,
                                            Qr, Kr, Vr, nullptr);
    dim3 gI(LI, NH);
    qkv_post_kernel<<<gI, 128, 0, stream>>>(qkvI, img_q_scale, img_k_scale, pe, LT, 1,
                                            Qr, Kr, Vr, Qimg);
    // V -> head-major transposed (H, HD, LL) for direct B-fragment loads
    dim3 gV(HD / 32, LL / 32, NH);
    transpose_u16_kernel<<<gV, dim3(32, 8), 0, stream>>>(Vr, Vtr, LL, HD);
  }

  // 6) main self-attention -> attn1 (LL, HID)
  {
    dim3 g(LL / 64, NH);
    attn_kernel<<<g, 128, 0, stream>>>(Qr, (long)LL * HD,
                                       Kr, HD, (long)LL * HD,
                                       Vtr, LL, (long)HD * LL,
                                       attn1, HID, HD, LL, LL, attn_scale);
  }

  // 7) IP k/v projections (M=16)
  {
    dim3 g(HID / BN, 1);
    gemm_kernel<<<g, 256, 0, stream>>>(image_proj, ip_k_w, ip_k_b, nullptr, nullptr,
                                       nullptr, nullptr, ipk, IPN, HID, CTX, 0);
    gemm_kernel<<<g, 256, 0, stream>>>(image_proj, ip_v_w, ip_v_b, nullptr, nullptr,
                                       nullptr, nullptr, ipv, IPN, HID, CTX, 0);
    int n = IPN * HID;
    cvt_kernel<<<(n + 255) / 256, 256, 0, stream>>>(ipk, ipkb, n);
    int m = NH * HD * IPPAD;
    ipv_transpose_kernel<<<(m + 255) / 256, 256, 0, stream>>>(ipv, ipvt);
  }

  // 8) IP attention: q = normalized (un-roped) img q -> ipattn (LI, HID)
  {
    dim3 g(LI / 64, NH);
    attn_kernel<<<g, 128, 0, stream>>>(Qimg, (long)LI * HD,
                                       ipkb, HID, (long)HD,
                                       ipvt, IPPAD, (long)HD * IPPAD,
                                       ipattn, HID, HD, LI, IPN, attn_scale);
  }

  // 9) output projections with gated residual (gate = mod[2])
  {
    dim3 gI(HID / BN, LI / BM);
    gemm_kernel<<<gI, 256, 0, stream>>>(attn1 + (size_t)LT * HID, img_proj_w, img_proj_b,
                                        img, modI + 2 * HID, nullptr, nullptr,
                                        img1, LI, HID, HID, 2);
    dim3 gT(HID / BN, LT / BM);
    gemm_kernel<<<gT, 256, 0, stream>>>(attn1, txt_proj_w, txt_proj_b,
                                        txt, modT + 2 * HID, nullptr, nullptr,
                                        txt1, LT, HID, HID, 2);
  }

  // 10) LN + mod (pre-MLP): sh=3, sc=4
  ln_mod_kernel<<<LI, 256, 0, stream>>>(img1, modI, 3, 4, lnI);
  ln_mod_kernel<<<LT, 256, 0, stream>>>(txt1, modT, 3, 4, lnT);

  // 11) MLP up + GELU
  {
    dim3 gI(MLPD / BN, LI / BM);
    gemm_kernel<<<gI, 256, 0, stream>>>(lnI, img_mlp_w1, img_mlp_b1, nullptr, nullptr,
                                        nullptr, nullptr, hI, LI, MLPD, HID, 1);
    dim3 gT(MLPD / BN, LT / BM);
    gemm_kernel<<<gT, 256, 0, stream>>>(lnT, txt_mlp_w1, txt_mlp_b1, nullptr, nullptr,
                                        nullptr, nullptr, hT, LT, MLPD, HID, 1);
  }

  // 12) MLP down + gated residual; img also fuses + ip_scale*ipattn (mode 3)
  {
    dim3 gI(HID / BN, LI / BM);
    gemm_kernel<<<gI, 256, 0, stream>>>(hI, img_mlp_w2, img_mlp_b2,
                                        img1, modI + 5 * HID, ipattn, ip_scale,
                                        OUT, LI, HID, MLPD, 3);
    dim3 gT(HID / BN, LT / BM);
    gemm_kernel<<<gT, 256, 0, stream>>>(hT, txt_mlp_w2, txt_mlp_b2,
                                        txt1, modT + 5 * HID, nullptr, nullptr,
                                        OUT + (size_t)LI * HID, LT, HID, MLPD, 2);
  }
}
